// Attention_57647051046942
// MI455X (gfx1250) — matrix operand order repaired and checked by reading, not compiled, not run on hardware
//
#include <hip/hip_runtime.h>

// ---------------------------------------------------------------------------
// Types / fragment helpers (CDNA5 wave32 WMMA, 16x16x32 f16 -> f32)
// ---------------------------------------------------------------------------
typedef _Float16 h16;
typedef __attribute__((ext_vector_type(16))) _Float16 v16h;
typedef __attribute__((ext_vector_type(8)))  _Float16 v8h;
typedef __attribute__((ext_vector_type(8)))  float    v8f;

union AFrag { v16h v; v8h h2[2]; h16 e[16]; };
union CFrag { v8f  v; float e[8]; };

// A fragment: 16x32 f16, rows m0..m0+15, cols k0..k0+31 of row-major A (lda).
// Lane l: half=l>>4, m=l&15.  Element e -> K = (e/8)*16 + half*8 + (e%8).
__device__ __forceinline__ v16h load_a_frag(const h16* A, int lda, int m0, int k0) {
  int lane = threadIdx.x & 31;
  int half = lane >> 4;
  int m    = lane & 15;
  const h16* p = A + (size_t)(m0 + m) * lda + k0 + half * 8;
  AFrag f;
  f.h2[0] = *(const v8h*)(p);        // K = k0 + half*8 + 0..7
  f.h2[1] = *(const v8h*)(p + 16);   // K = k0 + 16 + half*8 + 0..7
  return f.v;
}

// B fragment: 32x16 f16 tile of B, given Bt = B^T row-major (N x K, stride ldb).
// Lane l: col n=l&15, half=l>>4.  Element e -> K = k0 + (e/8)*16 + half*8 + (e%8).
__device__ __forceinline__ v16h load_b_frag(const h16* Bt, int ldb, int k0, int n0) {
  int lane = threadIdx.x & 31;
  int half = lane >> 4;
  int n    = lane & 15;
  const h16* p = Bt + (size_t)(n0 + n) * ldb + k0 + half * 8;
  AFrag f;
  f.h2[0] = *(const v8h*)(p);
  f.h2[1] = *(const v8h*)(p + 16);
  return f.v;
}

#define WMMA_F16(A_, B_, C_) \
  __builtin_amdgcn_wmma_f32_16x16x32_f16(false, (A_), false, (B_), (short)0, (C_), false, false)

// ---------------------------------------------------------------------------
// Elementwise conversion / layout kernels
// ---------------------------------------------------------------------------
__global__ void __launch_bounds__(256)
f32_to_f16_kernel(const float* __restrict__ in, h16* __restrict__ out, long n) {
  long i = (long)blockIdx.x * blockDim.x + threadIdx.x;
  if (i < n) out[i] = (h16)in[i];
}

// in: (Kd x Nd) row-major f32 -> out: (Nd x Kd) row-major f16  (W -> W^T)
__global__ void __launch_bounds__(256)
transpose_f32_to_f16_kernel(const float* __restrict__ in, h16* __restrict__ out,
                            int Kd, int Nd) {
  long i = (long)blockIdx.x * blockDim.x + threadIdx.x;
  if (i >= (long)Kd * Nd) return;
  int  n = (int)(i % Nd);
  long k = i / Nd;
  out[(size_t)n * Kd + k] = (h16)in[i];
}

// RoPE (adjacent-pair rotation) + head-major reorder:
// in (B,T,Hn,128) -> out (B,Hn,T,128).  cos/sin: (T,64).
__global__ void __launch_bounds__(256)
rope_reorder_kernel(const h16* __restrict__ in, h16* __restrict__ out,
                    const float* __restrict__ cosT, const float* __restrict__ sinT,
                    int T_, int Hn, long npairs) {
  long i = (long)blockIdx.x * blockDim.x + threadIdx.x;
  if (i >= npairs) return;
  int  p  = (int)(i & 63);
  long t1 = i >> 6;
  int  hh = (int)(t1 % Hn);
  long t2 = t1 / Hn;
  int  t  = (int)(t2 % T_);
  int  b  = (int)(t2 / T_);
  float c = cosT[t * 64 + p];
  float s = sinT[t * 64 + p];
  size_t src = (((size_t)b * T_ + t) * Hn + hh) * 128 + 2 * p;
  float x0 = (float)in[src];
  float x1 = (float)in[src + 1];
  size_t dst = (((size_t)b * Hn + hh) * T_ + t) * 128 + 2 * p;
  out[dst]     = (h16)(x0 * c - x1 * s);
  out[dst + 1] = (h16)(x0 * s + x1 * c);
}

// V: (B,T,8,128) -> V^T: (B,8,128,T)   (so PV B-fragments are contiguous in s)
__global__ void __launch_bounds__(256)
v_transpose_kernel(const h16* __restrict__ in, h16* __restrict__ out, int T_, long n) {
  long i = (long)blockIdx.x * blockDim.x + threadIdx.x;
  if (i >= n) return;
  int  d  = (int)(i & 127);
  long t1 = i >> 7;
  int  hh = (int)(t1 & 7);
  long t2 = t1 >> 3;
  int  t  = (int)(t2 % T_);
  int  b  = (int)(t2 / T_);
  out[(((size_t)b * 8 + hh) * 128 + d) * T_ + t] = in[i];
}

// ---------------------------------------------------------------------------
// WMMA GEMM: C(MxN) = A(MxK,f16) * B(KxN)  with Bt = B^T (NxK,f16) row-major.
// Register-blocked: one wave computes a 32x64 output block (2x4 accumulator
// tiles) so each 32-K step does 8 WMMAs on 6 fragment loads (~2.7x less L2
// traffic than 1 tile/wave).  global_prefetch_b8 hides next-step latency.
// ---------------------------------------------------------------------------
template <bool OUT_F32>
__global__ void __launch_bounds__(256)
wmma_gemm_kernel(const h16* __restrict__ A, const h16* __restrict__ Bt,
                 void* __restrict__ Cout, int M, int N, int K) {
  int wave = (int)((blockIdx.x * (unsigned)blockDim.x + threadIdx.x) >> 5);
  int lane = threadIdx.x & 31;
  int ntn  = N >> 6;                   // N / 64
  int mt   = wave / ntn;
  int nt   = wave - mt * ntn;
  if (mt >= (M >> 5)) return;          // uniform per wave (EXEC stays all-1s)
  int m0 = mt << 5, n0 = nt << 6;

  v8f acc[2][4];
#pragma unroll
  for (int i = 0; i < 2; ++i)
#pragma unroll
    for (int j = 0; j < 4; ++j) acc[i][j] = (v8f){};

  for (int k0 = 0; k0 < K; k0 += 32) {
    v16h a0 = load_a_frag(A, K, m0,      k0);
    v16h a1 = load_a_frag(A, K, m0 + 16, k0);
    if (k0 + 32 < K) {                 // prefetch next K-step (gfx1250 global_prefetch_b8)
      __builtin_prefetch(A  + (size_t)(m0 + lane) * K + k0 + 32, 0, 0);
      __builtin_prefetch(Bt + (size_t)(n0 + lane) * K + k0 + 32, 0, 0);
      __builtin_prefetch(Bt + (size_t)(n0 + 32 + lane) * K + k0 + 32, 0, 0);
    }
#pragma unroll
    for (int j = 0; j < 4; ++j) {
      v16h b = load_b_frag(Bt, K, k0, n0 + 16 * j);
      acc[0][j] = WMMA_F16(a0, b, acc[0][j]);
      acc[1][j] = WMMA_F16(a1, b, acc[1][j]);
    }
  }

  int half = lane >> 4, n = lane & 15;
#pragma unroll
  for (int i = 0; i < 2; ++i) {
#pragma unroll
    for (int j = 0; j < 4; ++j) {
      CFrag c; c.v = acc[i][j];
      if (OUT_F32) {
        float* C = (float*)Cout;
#pragma unroll
        for (int r = 0; r < 8; ++r)
          C[(size_t)(m0 + 16 * i + half * 8 + r) * N + n0 + 16 * j + n] = c.e[r];
      } else {
        h16* C = (h16*)Cout;
#pragma unroll
        for (int r = 0; r < 8; ++r)
          C[(size_t)(m0 + 16 * i + half * 8 + r) * N + n0 + 16 * j + n] = (h16)c.e[r];
      }
    }
  }
}

// ---------------------------------------------------------------------------
// Flash attention: one wave handles 16 query rows x HD=128 for one (b,h).
// q: (B,32,T,128)  k: (B,8,T,128)  vt: (B,8,128,T)  attn out: (B,T,4096) f16
// Online softmax over s in chunks of 32; P tile goes C-layout -> A-layout via
// a wave-private LDS slab (s_wait_dscnt to honor CDNA5 split counters).
// ---------------------------------------------------------------------------
__global__ void __launch_bounds__(256)
flash_attn_kernel(const h16* __restrict__ q, const h16* __restrict__ k,
                  const h16* __restrict__ vt, h16* __restrict__ attn, int T_) {
  constexpr int HD_ = 128, H_ = 32, HKV_ = 8;
  __shared__ h16 plds[8][16 * 32];
  int wib  = threadIdx.x >> 5;
  int lane = threadIdx.x & 31;
  int wave = blockIdx.x * 8 + wib;
  int qt = T_ >> 4;
  int bh = wave / qt;
  int m0 = (wave - bh * qt) << 4;
  int b  = bh / H_;
  int h  = bh - b * H_;
  int hk = h >> 2;                        // N_REP = 4
  const h16* Q  = q  + (size_t)(b * H_  + h)  * T_  * HD_;
  const h16* Kp = k  + (size_t)(b * HKV_ + hk) * T_  * HD_;
  const h16* Vt = vt + (size_t)(b * HKV_ + hk) * HD_ * T_;

  v16h qa[4];
#pragma unroll
  for (int j = 0; j < 4; ++j) qa[j] = load_a_frag(Q, HD_, m0, j * 32);

  CFrag o[8];
#pragma unroll
  for (int j = 0; j < 8; ++j)
#pragma unroll
    for (int r = 0; r < 8; ++r) o[j].e[r] = 0.f;

  float mrow[8], lrow[8];
#pragma unroll
  for (int r = 0; r < 8; ++r) { mrow[r] = -1e30f; lrow[r] = 0.f; }

  const float scale = 0.08838834764831845f;   // 1/sqrt(128)
  int halfw = lane >> 4, nl = lane & 15;
  h16* P = &plds[wib][0];

  for (int s0 = 0; s0 < T_; s0 += 32) {
    v8f slo = {}, shi = {};
#pragma unroll
    for (int j = 0; j < 4; ++j) {        // K-dim = HD = 4 x 32
      v16h b0 = load_b_frag(Kp, HD_, j * 32, s0);
      slo = WMMA_F16(qa[j], b0, slo);
      v16h b1 = load_b_frag(Kp, HD_, j * 32, s0 + 16);
      shi = WMMA_F16(qa[j], b1, shi);
    }
    CFrag SL, SH; SL.v = slo; SH.v = shi;
#pragma unroll
    for (int r = 0; r < 8; ++r) {        // row M = 8*halfw + r, col = s0 + nl / s0+16+nl
      float a = SL.e[r] * scale;
      float c = SH.e[r] * scale;
      float mx = fmaxf(a, c);
#pragma unroll
      for (int d = 1; d < 16; d <<= 1) mx = fmaxf(mx, __shfl_xor(mx, d, 32));
      float mnew = fmaxf(mrow[r], mx);
      float corr = __expf(mrow[r] - mnew);
      float pl = __expf(a - mnew);
      float ph = __expf(c - mnew);
      float rs = pl + ph;
#pragma unroll
      for (int d = 1; d < 16; d <<= 1) rs += __shfl_xor(rs, d, 32);
      lrow[r] = lrow[r] * corr + rs;
      mrow[r] = mnew;
#pragma unroll
      for (int j = 0; j < 8; ++j) o[j].e[r] *= corr;
      P[(halfw * 8 + r) * 32 + nl]      = (h16)pl;   // P tile, row-major 16x32
      P[(halfw * 8 + r) * 32 + 16 + nl] = (h16)ph;
    }
    asm volatile("s_wait_dscnt 0" ::: "memory");     // LDS stores visible to reads
    v16h pf = load_a_frag(P, 32, 0, 0);              // re-read P as A-fragment
#pragma unroll
    for (int j = 0; j < 8; ++j) {                    // O(16x128) += P(16x32) @ V
      v16h vb = load_b_frag(Vt, T_, s0, j * 16);
      o[j].v = WMMA_F16(pf, vb, o[j].v);
    }
  }
#pragma unroll
  for (int r = 0; r < 8; ++r) {
    float inv = 1.f / lrow[r];
    size_t row = (size_t)b * T_ + m0 + halfw * 8 + r;
#pragma unroll
    for (int j = 0; j < 8; ++j)
      attn[row * (size_t)(H_ * HD_) + h * HD_ + j * 16 + nl] = (h16)(o[j].e[r] * inv);
  }
}

// ---------------------------------------------------------------------------
// Host-side orchestration
// ---------------------------------------------------------------------------
extern "C" void kernel_launch(void* const* d_in, const int* in_sizes, int n_in,
                              void* d_out, int out_size, void* d_ws, size_t ws_size,
                              hipStream_t stream) {
  (void)in_sizes; (void)n_in; (void)out_size; (void)ws_size;
  const int Bn = 2, T_ = 2048, D_ = 4096, H_ = 32, HKV_ = 8, HD_ = 128;
  const int M  = Bn * T_;              // 4096
  const int NQ = H_ * HD_;             // 4096
  const int NK = HKV_ * HD_;           // 1024

  const float* x  = (const float*)d_in[0];
  const float* fc = (const float*)d_in[1];
  const float* fs = (const float*)d_in[2];
  const float* wq = (const float*)d_in[6];
  const float* wk = (const float*)d_in[7];
  const float* wv = (const float*)d_in[8];
  const float* wo = (const float*)d_in[9];

  char*  ws  = (char*)d_ws;
  size_t off = 0;
  auto alloc = [&](size_t bytes) -> char* {
    char* p = ws + off; off += (bytes + 255) & ~(size_t)255; return p;
  };
  h16* xh  = (h16*)alloc((size_t)M  * D_ * 2);   // x, f16
  h16* wqT = (h16*)alloc((size_t)D_ * NQ * 2);   // wq^T
  h16* wkT = (h16*)alloc((size_t)D_ * NK * 2);
  h16* wvT = (h16*)alloc((size_t)D_ * NK * 2);
  h16* woT = (h16*)alloc((size_t)NQ * D_ * 2);   // wo^T
  h16* qh  = (h16*)alloc((size_t)M * NQ * 2);    // pre-RoPE Q  (B,T,H,HD)
  h16* kh  = (h16*)alloc((size_t)M * NK * 2);    // pre-RoPE K  (B,T,HKV,HD)
  h16* vh  = (h16*)alloc((size_t)M * NK * 2);    // V           (B,T,HKV,HD)
  h16* qr  = (h16*)alloc((size_t)M * NQ * 2);    // Q roped     (B,H,T,HD)
  h16* kr  = (h16*)alloc((size_t)M * NK * 2);    // K roped     (B,HKV,T,HD)
  h16* vtp = (h16*)alloc((size_t)M * NK * 2);    // V^T         (B,HKV,HD,T)
  h16* att = (h16*)alloc((size_t)M * NQ * 2);    // attention out (B,T,H*HD)

  const int TPB = 256;
  // 1) fp32 -> f16 conversions (weights transposed to B^T layout)
  f32_to_f16_kernel<<<(long)M * D_ / TPB, TPB, 0, stream>>>(x, xh, (long)M * D_);
  transpose_f32_to_f16_kernel<<<(long)D_ * NQ / TPB, TPB, 0, stream>>>(wq, wqT, D_, NQ);
  transpose_f32_to_f16_kernel<<<(long)D_ * NK / TPB, TPB, 0, stream>>>(wk, wkT, D_, NK);
  transpose_f32_to_f16_kernel<<<(long)D_ * NK / TPB, TPB, 0, stream>>>(wv, wvT, D_, NK);
  transpose_f32_to_f16_kernel<<<(long)NQ * D_ / TPB, TPB, 0, stream>>>(wo, woT, NQ, D_);

  // 2) QKV projections (WMMA, f32 accumulate, f16 out); 32x64 block per wave
  wmma_gemm_kernel<false><<<(M / 32) * (NQ / 64) / 8, TPB, 0, stream>>>(xh, wqT, qh, M, NQ, D_);
  wmma_gemm_kernel<false><<<(M / 32) * (NK / 64) / 8, TPB, 0, stream>>>(xh, wkT, kh, M, NK, D_);
  wmma_gemm_kernel<false><<<(M / 32) * (NK / 64) / 8, TPB, 0, stream>>>(xh, wvT, vh, M, NK, D_);

  // 3) RoPE + head-major reorder; V transpose for PV B-fragments
  rope_reorder_kernel<<<(long)M * H_   * 64 / TPB, TPB, 0, stream>>>(qh, qr, fc, fs, T_, H_,   (long)M * H_   * 64);
  rope_reorder_kernel<<<(long)M * HKV_ * 64 / TPB, TPB, 0, stream>>>(kh, kr, fc, fs, T_, HKV_, (long)M * HKV_ * 64);
  v_transpose_kernel<<<(long)M * NK / TPB, TPB, 0, stream>>>(vh, vtp, T_, (long)M * NK);

  // 4) Flash attention (one wave per 16-query tile per head)
  flash_attn_kernel<<<Bn * H_ * (T_ / 16) / 8, TPB, 0, stream>>>(qr, kr, vtp, att, T_);

  // 5) Output projection -> fp32 d_out
  wmma_gemm_kernel<true><<<(M / 32) * (D_ / 64) / 8, TPB, 0, stream>>>(att, woT, d_out, M, D_, NQ);
}
